// GroupCRouter_78288663872361
// MI455X (gfx1250) — matrix operand order repaired and checked by reading, not compiled, not run on hardware
//
#include <hip/hip_runtime.h>
#include <hip/hip_bf16.h>

// ---- CDNA5 (gfx1250) wave32 WMMA types ----
typedef __attribute__((ext_vector_type(16))) __bf16 v16bf;
typedef __attribute__((ext_vector_type(8)))  __bf16 v8bf;
typedef __attribute__((ext_vector_type(8)))  float  v8f;
typedef __attribute__((ext_vector_type(4)))  float  v4f;
typedef __attribute__((ext_vector_type(2)))  float  v2f;

#define Bb   4096
#define Nn   128
#define Dd   256
#define Ff   64      // D/4
#define Ee   6       // experts
#define TM   128     // tokens per block (== Nn, so blockIdx.x == batch index)

// round-to-nearest-even f32 -> bf16 (pure integer ops; LDS holds raw u16 bits)
__device__ __forceinline__ unsigned int pack2bf(float x, float y) {
    unsigned int ux = __float_as_uint(x);
    ux += 0x7FFFu + ((ux >> 16) & 1u);
    unsigned int uy = __float_as_uint(y);
    uy += 0x7FFFu + ((uy >> 16) & 1u);
    return (ux >> 16) | (uy & 0xFFFF0000u);
}
__device__ __forceinline__ unsigned short f2bf(float f) {
    unsigned int u = __float_as_uint(f);
    u += 0x7FFFu + ((u >> 16) & 1u);
    return (unsigned short)(u >> 16);
}

__device__ __forceinline__ float gelu_exact(float x) {
    return 0.5f * x * (1.0f + erff(x * 0.70710678118654752f));
}

#define WMMA_BF16(A, B, C) \
    __builtin_amdgcn_wmma_f32_16x16x32_bf16(false, (A), false, (B), (short)0, (C), false, false)

__global__ __launch_bounds__(256) void
router_wmma_kernel(const float* __restrict__ tokens,
                   const int*   __restrict__ token_types,
                   const int*   __restrict__ t,
                   const float* __restrict__ base_assignment,
                   const float* __restrict__ W1,
                   const float* __restrict__ b1,
                   const float* __restrict__ W2,
                   const float* __restrict__ b2,
                   float* __restrict__ out) {
    // +8 bf16 row padding to de-conflict strided fragment reads (row stride 528 B)
    __shared__ unsigned short Xs [TM][Dd + 8];   // 67.5 KB  tokens tile, bf16
    __shared__ unsigned short W1t[Ff][Dd + 8];   // 33   KB  W1 transposed (N-major)
    __shared__ unsigned short Hs [TM][Ff + 8];   // 18   KB  gelu(X@W1+b1), bf16
    __shared__ unsigned short W2t[16][Ff + 8];   //  2.3 KB  W2 transposed, padded N->16
    __shared__ float          Lg [TM][9];        //  4.5 KB  scaled logits
    __shared__ float          b1s[Ff];
    __shared__ float          bas[40];           // base_assignment (6x6)

    const int tid = threadIdx.x;
    const int blk = blockIdx.x;

    // ------------- Phase 0: cooperative loads + f32->bf16 conversion -------------
    {
        // tokens: single-pass 512 MB stream -> non-temporal so L2 keeps W1/W2 hot
        const v4f* src = (const v4f*)(tokens + (size_t)blk * TM * Dd);
        for (int idx = tid; idx < TM * Dd / 4; idx += 256) {
            v4f v = __builtin_nontemporal_load(src + idx);
            int row = idx >> 6;          // 64 float4 per 256-wide row
            int c   = (idx & 63) << 2;
            uint2 pk;
            pk.x = pack2bf(v.x, v.y);
            pk.y = pack2bf(v.z, v.w);
            *(uint2*)&Xs[row][c] = pk;   // 528-B row stride, c%4==0 -> 8B aligned
        }
    }
    for (int idx = tid; idx < Dd * Ff; idx += 256) {   // W1 is (D,F) row-major
        int d = idx >> 6;
        int f = idx & 63;
        W1t[f][d] = f2bf(W1[idx]);                      // transpose -> N-major
    }
    for (int idx = tid; idx < 16 * Ff; idx += 256) {    // W2 is (F,E) row-major
        int n = idx >> 6;
        int k = idx & 63;
        W2t[n][k] = (n < Ee) ? f2bf(W2[k * Ee + n]) : (unsigned short)0;
    }
    if (tid < Ff) b1s[tid] = b1[tid];
    if (tid < Ee * Ee) bas[tid] = base_assignment[tid];
    __syncthreads();

    const int wave = tid >> 5;
    const int lane = tid & 31;
    const int lr   = lane & 15;    // A row-within-tile / B,C column
    const int hh   = lane >> 4;    // lane half selects K sub-range / M sub-range

    // ------------- Phase 1: H = gelu(X @ W1 + b1) via v_wmma_f32_16x16x32_bf16 ----
    // 2M x 2N tile per wave: 4 fragment loads feed 4 WMMAs (vs 5 with 1Mx4N),
    // software-pipelined one kt ahead to hide ds_load latency behind WMMA issue.
    {
        const int mg = wave >> 1;            // 0..3 -> M rows [mg*32, mg*32+32)
        const int ng = wave & 1;             // 0..1 -> N cols [ng*32, ng*32+32)
        const int rowA0 = mg * 32 + lr;
        const int rowA1 = mg * 32 + 16 + lr;
        const int colB0 = ng * 32 + lr;
        const int colB1 = ng * 32 + 16 + lr;

        auto loadA = [&](int kt, int row) -> v16bf {
            const int kb = kt * 32 + hh * 16; // half 0: K 0..15, half 1: K 16..31
            v8bf lo = *(const v8bf*)&Xs[row][kb];
            v8bf hi = *(const v8bf*)&Xs[row][kb + 8];
            return __builtin_shufflevector(lo, hi,
                     0,1,2,3,4,5,6,7,8,9,10,11,12,13,14,15);
        };
        auto loadB = [&](int kt, int col) -> v16bf {
            const int kb = kt * 32 + hh * 16; // lane = N, contiguous 16 K per lane half
            return *(const v16bf*)&W1t[col][kb];
        };

        v8f acc00 = {}, acc01 = {}, acc10 = {}, acc11 = {};
        v16bf a0 = loadA(0, rowA0), a1 = loadA(0, rowA1);
        v16bf b0 = loadB(0, colB0), b1f = loadB(0, colB1);

        #pragma unroll
        for (int kt = 0; kt < Dd / 32; ++kt) {
            v16bf na0 = a0, na1 = a1, nb0 = b0, nb1 = b1f;
            if (kt < Dd / 32 - 1) {          // prefetch next-kt fragments
                na0 = loadA(kt + 1, rowA0);
                na1 = loadA(kt + 1, rowA1);
                nb0 = loadB(kt + 1, colB0);
                nb1 = loadB(kt + 1, colB1);
            }
            acc00 = WMMA_BF16(a0, b0,  acc00);
            acc01 = WMMA_BF16(a0, b1f, acc01);
            acc10 = WMMA_BF16(a1, b0,  acc10);
            acc11 = WMMA_BF16(a1, b1f, acc11);
            a0 = na0; a1 = na1; b0 = nb0; b1f = nb1;
        }

        // C layout: VGPR j -> M = j + 8*hh, lane -> N; apply bias + exact GELU
        const int r0 = mg * 32 + hh * 8;          // acc0x rows
        const int r1 = mg * 32 + 16 + hh * 8;     // acc1x rows
        const int c0 = ng * 32 + lr;              // accx0 col
        const int c1 = ng * 32 + 16 + lr;         // accx1 col
        const float bia0 = b1s[c0], bia1 = b1s[c1];
        #pragma unroll
        for (int j = 0; j < 8; ++j) {
            Hs[r0 + j][c0] = f2bf(gelu_exact(acc00[j] + bia0));
            Hs[r0 + j][c1] = f2bf(gelu_exact(acc01[j] + bia1));
            Hs[r1 + j][c0] = f2bf(gelu_exact(acc10[j] + bia0));
            Hs[r1 + j][c1] = f2bf(gelu_exact(acc11[j] + bia1));
        }
    }
    __syncthreads();

    // ------------- Phase 2: logits = (H @ W2 + b2) * TEMP --------------------------
    {
        v8f acc = {};
        const int arow = wave * 16 + lr;
        #pragma unroll
        for (int kt = 0; kt < Ff / 32; ++kt) {
            const int kb = kt * 32 + hh * 16;
            v8bf lo = *(const v8bf*)&Hs[arow][kb];
            v8bf hi = *(const v8bf*)&Hs[arow][kb + 8];
            v16bf a = __builtin_shufflevector(lo, hi,
                        0,1,2,3,4,5,6,7,8,9,10,11,12,13,14,15);
            const int kbB = kt * 32 + hh * 16;
            v16bf b = *(const v16bf*)&W2t[lr][kbB];
            acc = WMMA_BF16(a, b, acc);
        }
        const int orow = wave * 16 + hh * 8;
        if (lr < Ee) {
            const float bb2 = b2[lr];
            #pragma unroll
            for (int j = 0; j < 8; ++j)
                Lg[orow + j][lr] = (acc[j] + bb2) * 0.1f;
        }
    }
    __syncthreads();

    // ------------- Phase 3: routing epilogue (1 token / thread) --------------------
    if (tid < TM) {
        const size_t m = (size_t)blk * TM + tid;
        const int type = token_types[m];

        float p[Ee];
        if (type == 5) {                      // TTYPE_UNKNOWN -> learned soft gate
            float lg[Ee], mx = -1e30f;
            #pragma unroll
            for (int e = 0; e < Ee; ++e) { lg[e] = Lg[tid][e]; mx = fmaxf(mx, lg[e]); }
            float s = 0.f;
            #pragma unroll
            for (int e = 0; e < Ee; ++e) { lg[e] = expf(lg[e] - mx); s += lg[e]; }
            const float inv = 1.f / s;
            #pragma unroll
            for (int e = 0; e < Ee; ++e) p[e] = lg[e] * inv;
        } else {                              // hard structural assignment
            #pragma unroll
            for (int e = 0; e < Ee; ++e) p[e] = bas[type * Ee + e];
        }

        // routing floor: alpha = FLOOR*E = 0.225
        #pragma unroll
        for (int e = 0; e < Ee; ++e) p[e] = 0.775f * p[e] + 0.0375f;

        // top-1 (ties -> lowest index, matching lax.top_k)
        int best = 0; float bv = p[0];
        #pragma unroll
        for (int e = 1; e < Ee; ++e) if (p[e] > bv) { bv = p[e]; best = e; }

        // per-batch capacity with redistribution
        const float cap = 0.5f + 0.1f * (float)t[blk] * (1.0f / 1000.0f);
        float d[Ee], hr[Ee];
        float sumEx = 0.f, hs = 0.f;
        #pragma unroll
        for (int e = 0; e < Ee; ++e) {
            float de = (e == best) ? bv : 0.f;
            float ex = fmaxf(de - cap, 0.f);
            de -= ex;  sumEx += ex;
            float h = fmaxf(cap - de, 0.f);
            hr[e] = h; hs += h;
            d[e] = de;
        }
        hs = fmaxf(hs, 1e-8f);
        float sumc = 0.f;
        #pragma unroll
        for (int e = 0; e < Ee; ++e) { d[e] += sumEx * hr[e] / hs; sumc += d[e]; }
        const float invc = 1.f / (sumc + 1e-8f);

        // outputs are write-once streams -> non-temporal float2 stores (24 B rows, 8B aligned)
        v2f* outD = (v2f*)(out + m * Ee);                          // dispatch_out
        v2f* outC = (v2f*)(out + (size_t)Bb * Nn * Ee + m * Ee);   // combine
        #pragma unroll
        for (int e = 0; e < Ee / 2; ++e) {
            v2f vd; vd.x = d[2*e];          vd.y = d[2*e + 1];
            v2f vc; vc.x = d[2*e] * invc;   vc.y = d[2*e + 1] * invc;
            __builtin_nontemporal_store(vd, outD + e);
            __builtin_nontemporal_store(vc, outC + e);
        }
    }
}

extern "C" void kernel_launch(void* const* d_in, const int* in_sizes, int n_in,
                              void* d_out, int out_size, void* d_ws, size_t ws_size,
                              hipStream_t stream) {
    const float* tokens          = (const float*)d_in[0];
    const int*   token_types     = (const int*)  d_in[1];
    const int*   t               = (const int*)  d_in[2];
    const float* base_assignment = (const float*)d_in[3];
    const float* W1              = (const float*)d_in[4];
    const float* b1              = (const float*)d_in[5];
    const float* W2              = (const float*)d_in[6];
    const float* b2              = (const float*)d_in[7];
    float* out = (float*)d_out;

    dim3 grid(Bb * Nn / TM);   // 4096 blocks, one batch row each
    dim3 block(256);           // 8 wave32s
    hipLaunchKernelGGL(router_wmma_kernel, grid, block, 0, stream,
                       tokens, token_types, t, base_assignment, W1, b1, W2, b2, out);
}
